// SwinBlock3D_20169166422854
// MI455X (gfx1250) — compile-verified, matrix operand order checked by reading, not run on hardware
//
#include <hip/hip_runtime.h>
#include <hip/hip_bf16.h>
#include <hip/hip_fp16.h>

// ---------------------------------------------------------------------------
// Swin-3D block on MI455X (gfx1250), wave32 + v_wmma_f32_16x16x32_f16.
//   B=2 D=8 H=128 W=128 C=96, heads=3, head_dim=32, window=(2,8,8) -> N=128,
//   NWIN=1024/batch, MLP=384, shift=(1,4,4).
// All GEMMs mapped to 16x16x32 f16 WMMA; intermediates kept f16 in ws.
// K/V tiles staged into LDS with gfx1250 async global->LDS copies.
// ---------------------------------------------------------------------------

typedef __attribute__((ext_vector_type(16))) _Float16 v16h;
typedef __attribute__((ext_vector_type(8)))  float    v8f;

#define DEV static __device__ __forceinline__

DEV v8f wmma32(v16h a, v16h b, v8f c) {
  // D = A(16x32 f16) x B(32x16 f16) + C(16x16 f32)
  return __builtin_amdgcn_wmma_f32_16x16x32_f16(
      /*neg_a=*/false, a, /*neg_b=*/false, b,
      /*c_mod=*/(short)0, c, /*reuse_a=*/false, /*reuse_b=*/false);
}

// A-fragment (16x32, row-major source, leading dim ld in elements).
// ISA 7.12.2: lane L holds row L%16; lanes 0-15 K-halves {0..7,16..23},
// lanes 16-31 K-halves {8..15,24..31}; vgpr v<4 -> k=2v, v>=4 -> k=2v+8.
DEV v16h frag_a_f16(const _Float16* p, int ld) {
  const int lane = threadIdx.x & 31;
  const _Float16* r = p + (size_t)(lane & 15) * ld + ((lane >> 4) & 1) * 8;
  v16h a;
#pragma unroll
  for (int v = 0; v < 4; ++v) { a[2 * v] = r[2 * v]; a[2 * v + 1] = r[2 * v + 1]; }
#pragma unroll
  for (int v = 4; v < 8; ++v) { a[2 * v] = r[2 * v + 8]; a[2 * v + 1] = r[2 * v + 9]; }
  return a;
}

// B-fragment (32x16, row-major K x N source): lane holds column N=lane%16,
// same K ordering as A.
DEV v16h frag_b_f16(const _Float16* p, int ld) {
  const int lane = threadIdx.x & 31;
  const _Float16* c = p + (lane & 15) + (size_t)(((lane >> 4) & 1) * 8) * ld;
  v16h b;
#pragma unroll
  for (int v = 0; v < 4; ++v) { b[2 * v] = c[(size_t)(2 * v) * ld]; b[2 * v + 1] = c[(size_t)(2 * v + 1) * ld]; }
#pragma unroll
  for (int v = 4; v < 8; ++v) { b[2 * v] = c[(size_t)(2 * v + 8) * ld]; b[2 * v + 1] = c[(size_t)(2 * v + 9) * ld]; }
  return b;
}

// B-fragment from f32 weights with on-the-fly conversion.
DEV v16h frag_b_f32(const float* p, int ld) {
  const int lane = threadIdx.x & 31;
  const float* c = p + (lane & 15) + (size_t)(((lane >> 4) & 1) * 8) * ld;
  v16h b;
#pragma unroll
  for (int v = 0; v < 4; ++v) { b[2 * v] = (_Float16)c[(size_t)(2 * v) * ld]; b[2 * v + 1] = (_Float16)c[(size_t)(2 * v + 1) * ld]; }
#pragma unroll
  for (int v = 4; v < 8; ++v) { b[2 * v] = (_Float16)c[(size_t)(2 * v + 8) * ld]; b[2 * v + 1] = (_Float16)c[(size_t)(2 * v + 9) * ld]; }
  return b;
}

// gfx1250 async global->LDS 16-byte copy (GV mode: vdst = LDS byte offset,
// vaddr = 64-bit global address).  Tracked by ASYNCcnt.
DEV void async_copy_b128(unsigned lds_off, const void* gaddr) {
  asm volatile("global_load_async_to_lds_b128 %0, %1, off"
               :: "v"(lds_off), "v"((unsigned long long)(uintptr_t)gaddr)
               : "memory");
}
DEV void async_wait0() {
  asm volatile("s_wait_asynccnt 0" ::: "memory");
}

// ---------------------------------------------------------------------------
// Kernel 1: LayerNorm + shift-roll + window partition -> Xn f16 [2048][128][96]
// One wave per token; wave32 shuffle reduction for mean/var.
// ---------------------------------------------------------------------------
__global__ __launch_bounds__(128) void k_ln_part(
    const float* __restrict__ x, const float* __restrict__ g,
    const float* __restrict__ bt, _Float16* __restrict__ xn) {
  const int wv = threadIdx.x >> 5;
  const int lane = threadIdx.x & 31;
  const int gtok = blockIdx.x * 4 + wv;  // 0..262143
  const int win = gtok >> 7;
  const int t = gtok & 127;
  const int b = win >> 10, wloc = win & 1023;
  const int dwi = wloc >> 8, hwi = (wloc >> 4) & 15, wwi = wloc & 15;
  const int td = t >> 6, th = (t >> 3) & 7, tw = t & 7;
  const int id = (dwi * 2 + td + 1) & 7;       // roll by -shift == read at p+shift
  const int ih = (hwi * 8 + th + 4) & 127;
  const int iw = (wwi * 8 + tw + 4) & 127;
  const float* xr = x + ((((size_t)(b * 8 + id)) * 128 + ih) * 128 + iw) * 96;
  float v0 = xr[lane], v1 = xr[lane + 32], v2 = xr[lane + 64];
  float s = v0 + v1 + v2, ss = v0 * v0 + v1 * v1 + v2 * v2;
#pragma unroll
  for (int off = 16; off; off >>= 1) {
    s += __shfl_xor(s, off, 32);
    ss += __shfl_xor(ss, off, 32);
  }
  float mu = s * (1.f / 96.f);
  float var = ss * (1.f / 96.f) - mu * mu;
  float rs = rsqrtf(var + 1e-5f);
  _Float16* o = xn + (size_t)gtok * 96;
  o[lane]      = (_Float16)((v0 - mu) * rs * g[lane]      + bt[lane]);
  o[lane + 32] = (_Float16)((v1 - mu) * rs * g[lane + 32] + bt[lane + 32]);
  o[lane + 64] = (_Float16)((v2 - mu) * rs * g[lane + 64] + bt[lane + 64]);
}

// ---------------------------------------------------------------------------
// Kernel 2: QKV GEMM [262144 x 96] x [96 x 288] -> f16.  One window (128 rows)
// per WG; 144 output tiles over 4 waves; K = 3 WMMA steps.
// ---------------------------------------------------------------------------
__global__ __launch_bounds__(128) void k_qkv(
    const _Float16* __restrict__ xn, const float* __restrict__ wqkv,
    _Float16* __restrict__ qkv) {
  const int wv = threadIdx.x >> 5;
  const int lane = threadIdx.x & 31;
  const size_t row0 = (size_t)blockIdx.x * 128;
#pragma unroll 1
  for (int tile = wv; tile < 144; tile += 4) {
    const int mt = tile / 18, nt = tile % 18;
    const _Float16* ap = xn + (row0 + mt * 16) * 96;
    __builtin_prefetch(ap, 0, 1);  // global_prefetch_b8
    v8f acc = {};
#pragma unroll
    for (int ks = 0; ks < 3; ++ks) {
      v16h a = frag_a_f16(ap + ks * 32, 96);
      v16h b = frag_b_f32(wqkv + (size_t)(ks * 32) * 288 + nt * 16, 288);
      acc = wmma32(a, b, acc);
    }
    const int col = nt * 16 + (lane & 15);
    const int rb = mt * 16 + ((lane >> 4) & 1) * 8;
#pragma unroll
    for (int r = 0; r < 8; ++r)
      qkv[(row0 + rb + r) * 288 + col] = (_Float16)acc[r];
  }
}

// ---------------------------------------------------------------------------
// Kernel 3: attention for one (window, head).
//  * K/V head-slices staged to LDS ONCE per WG via async global->LDS b128
//    copies (each QKV head slice is 64B/row, 16B aligned) -> 4x less global
//    traffic than per-wave gathers, overlapped with Q loads + relb fill.
//  * Scores (32 rows x 128) live in registers (16 v8f/wave); softmax via
//    shfl_xor inside the 16-lane column groups of the WMMA C layout.
//  * P staged through per-wave LDS for the A-fragment re-swizzle of P.V.
// ---------------------------------------------------------------------------
__global__ __launch_bounds__(128, 1) void k_attn(
    const _Float16* __restrict__ qkv, const float* __restrict__ rel_bias,
    _Float16* __restrict__ aout) {
  __shared__ float relb[2025];           // 675 x 3 heads
  __shared__ _Float16 Ks[128][32];       // K head slice
  __shared__ _Float16 Vs[128][32];       // V head slice
  __shared__ _Float16 P[4][32][136];     // per-wave prob tiles (pad 8)
  const int win = blockIdx.x / 3;
  const int head = blockIdx.x % 3;
  const size_t base = (size_t)win * 128 * 288;

  // -- async-stage K,V: 512 b128 transfers each, 4 per row --
  {
    const unsigned kbase = (unsigned)(uintptr_t)&Ks[0][0];
    const unsigned vbase = (unsigned)(uintptr_t)&Vs[0][0];
    const _Float16* gk = qkv + base + 96 + head * 32;
    const _Float16* gv = qkv + base + 192 + head * 32;
#pragma unroll
    for (int it = 0; it < 4; ++it) {
      const int idx = threadIdx.x + it * 128;  // transfer id 0..511
      const int row = idx >> 2, seg = idx & 3; // 16B segment within 64B row
      const unsigned loff = (unsigned)(row * 64 + seg * 16);
      async_copy_b128(kbase + loff, gk + (size_t)row * 288 + seg * 8);
      async_copy_b128(vbase + loff, gv + (size_t)row * 288 + seg * 8);
    }
  }
  for (int i = threadIdx.x; i < 2025; i += 128) relb[i] = rel_bias[i];

  const int wv = threadIdx.x >> 5, lane = threadIdx.x & 31;
  const int half = (lane >> 4) & 1, col = lane & 15;
  const int r0 = wv * 32;
  const _Float16* Q = qkv + base + head * 32;

  // Q fragments can be fetched while the async engine fills LDS.
  v16h aq0 = frag_a_f16(Q + (size_t)r0 * 288, 288);
  v16h aq1 = frag_a_f16(Q + (size_t)(r0 + 16) * 288, 288);

  async_wait0();      // ASYNCcnt == 0: this wave's K/V transfers landed
  __syncthreads();    // make all waves' transfers + relb visible

  // ---- S = Q K^T : K^T B-fragments use the A-gather off row-major Ks ----
  v8f s[2][8];
#pragma unroll
  for (int nt = 0; nt < 8; ++nt) {
    v16h bk = frag_a_f16(&Ks[nt * 16][0], 32);
    v8f z0 = {}; s[0][nt] = wmma32(aq0, bk, z0);
    v8f z1 = {}; s[1][nt] = wmma32(aq1, bk, z1);
  }

  // ---- scale + relative-position bias + shifted-window region mask ----
  const int wloc = win & 1023;
  const int dwi = wloc >> 8, hwi = (wloc >> 4) & 15, wwi = wloc & 15;
  int jreg[8], jd[8], jh[8], jw[8];
#pragma unroll
  for (int nt = 0; nt < 8; ++nt) {
    int j = nt * 16 + col;
    int tdj = j >> 6, thj = (j >> 3) & 7, twj = j & 7;
    jd[nt] = tdj; jh[nt] = thj; jw[nt] = twj;
    int pd = dwi * 2 + tdj, ph = hwi * 8 + thj, pw = wwi * 8 + twj;
    int bd = (pd < 6) ? 0 : (pd < 7 ? 1 : 2);
    int bh = (ph < 120) ? 0 : (ph < 124 ? 1 : 2);
    int bw = (pw < 120) ? 0 : (pw < 124 ? 1 : 2);
    jreg[nt] = bd * 9 + bh * 3 + bw;
  }
  const float scale = 0.17677669529663687f;  // 32^-0.5
#pragma unroll
  for (int mt = 0; mt < 2; ++mt) {
#pragma unroll
    for (int r = 0; r < 8; ++r) {
      int i = r0 + mt * 16 + half * 8 + r;
      int tdi = i >> 6, thi = (i >> 3) & 7, twi = i & 7;
      int pd = dwi * 2 + tdi, ph = hwi * 8 + thi, pw = wwi * 8 + twi;
      int bd = (pd < 6) ? 0 : (pd < 7 ? 1 : 2);
      int bh = (ph < 120) ? 0 : (ph < 124 ? 1 : 2);
      int bw = (pw < 120) ? 0 : (pw < 124 ? 1 : 2);
      int ireg = bd * 9 + bh * 3 + bw;
#pragma unroll
      for (int nt = 0; nt < 8; ++nt) {
        int idx = (tdi - jd[nt] + 1) * 225 + (thi - jh[nt] + 7) * 15 + (twi - jw[nt] + 7);
        float m = (ireg != jreg[nt]) ? -1e9f : 0.f;
        s[mt][nt][r] = s[mt][nt][r] * scale + relb[idx * 3 + head] + m;
      }
    }
  }

  // ---- softmax over 128 cols; row lives inside one 16-lane half group ----
#pragma unroll
  for (int mt = 0; mt < 2; ++mt) {
#pragma unroll
    for (int r = 0; r < 8; ++r) {
      float mx = -3.4e38f;
#pragma unroll
      for (int nt = 0; nt < 8; ++nt) mx = fmaxf(mx, s[mt][nt][r]);
#pragma unroll
      for (int off = 1; off < 16; off <<= 1) mx = fmaxf(mx, __shfl_xor(mx, off, 32));
      float sum = 0.f;
#pragma unroll
      for (int nt = 0; nt < 8; ++nt) {
        float e = __expf(s[mt][nt][r] - mx);
        s[mt][nt][r] = e;
        sum += e;
      }
#pragma unroll
      for (int off = 1; off < 16; off <<= 1) sum += __shfl_xor(sum, off, 32);
      float inv = 1.f / sum;
#pragma unroll
      for (int nt = 0; nt < 8; ++nt) s[mt][nt][r] *= inv;
    }
  }

  // ---- stage P (f16) per-wave in LDS in row-major order ----
#pragma unroll
  for (int mt = 0; mt < 2; ++mt)
#pragma unroll
    for (int nt = 0; nt < 8; ++nt)
#pragma unroll
      for (int r = 0; r < 8; ++r)
        P[wv][mt * 16 + half * 8 + r][nt * 16 + col] = (_Float16)s[mt][nt][r];
  __syncthreads();

  // ---- O = P V  (K = 128 tokens, 4 WMMA steps; Vs row-major [tok][dim]) ----
  v8f o00 = {}, o01 = {}, o10 = {}, o11 = {};
#pragma unroll
  for (int ks = 0; ks < 4; ++ks) {
    v16h b0 = frag_b_f16(&Vs[ks * 32][0], 32);
    v16h b1 = frag_b_f16(&Vs[ks * 32][16], 32);
    v16h a0 = frag_a_f16(&P[wv][0][ks * 32], 136);
    v16h a1 = frag_a_f16(&P[wv][16][ks * 32], 136);
    o00 = wmma32(a0, b0, o00);
    o01 = wmma32(a0, b1, o01);
    o10 = wmma32(a1, b0, o10);
    o11 = wmma32(a1, b1, o11);
  }
  const int dc0 = head * 32 + col, dc1 = head * 32 + 16 + col;
#pragma unroll
  for (int r = 0; r < 8; ++r) {
    size_t t0 = (size_t)win * 128 + r0 + half * 8 + r;
    size_t t1 = t0 + 16;
    aout[t0 * 96 + dc0] = (_Float16)o00[r];
    aout[t0 * 96 + dc1] = (_Float16)o01[r];
    aout[t1 * 96 + dc0] = (_Float16)o10[r];
    aout[t1 * 96 + dc1] = (_Float16)o11[r];
  }
}

// ---------------------------------------------------------------------------
// Kernel 4: out-projection + bias + residual, inverse roll folded into the
// scatter.  w_out staged f16 in LDS; one window per WG.
// ---------------------------------------------------------------------------
__global__ __launch_bounds__(128) void k_proj(
    const _Float16* __restrict__ aout, const float* __restrict__ wout,
    const float* __restrict__ bout, const float* __restrict__ x,
    float* __restrict__ y) {
  __shared__ _Float16 Wl[96][96];
  __shared__ float bl[96];
  for (int i = threadIdx.x; i < 96 * 96; i += 128)
    Wl[i / 96][i % 96] = (_Float16)wout[i];
  if (threadIdx.x < 96) bl[threadIdx.x] = bout[threadIdx.x];
  __syncthreads();
  const int wv = threadIdx.x >> 5, lane = threadIdx.x & 31;
  const int half = (lane >> 4) & 1, cl = lane & 15;
  const int win = blockIdx.x;
  const int b = win >> 10, wloc = win & 1023;
  const int dwi = wloc >> 8, hwi = (wloc >> 4) & 15, wwi = wloc & 15;
  const size_t row0 = (size_t)win * 128;
#pragma unroll 1
  for (int tile = wv; tile < 48; tile += 4) {
    const int mt = tile / 6, nt = tile % 6;
    v8f acc = {};
#pragma unroll
    for (int ks = 0; ks < 3; ++ks) {
      v16h a = frag_a_f16(aout + (row0 + mt * 16) * 96 + ks * 32, 96);
      v16h bf = frag_b_f16(&Wl[ks * 32][nt * 16], 96);
      acc = wmma32(a, bf, acc);
    }
    const int col = nt * 16 + cl;
#pragma unroll
    for (int r = 0; r < 8; ++r) {
      int t = mt * 16 + half * 8 + r;
      int td = t >> 6, th = (t >> 3) & 7, tw = t & 7;
      int id = (dwi * 2 + td + 1) & 7;
      int ih = (hwi * 8 + th + 4) & 127;
      int iw = (wwi * 8 + tw + 4) & 127;
      size_t addr = ((((size_t)(b * 8 + id)) * 128 + ih) * 128 + iw) * 96 + col;
      y[addr] = x[addr] + acc[r] + bl[col];
    }
  }
}

// ---------------------------------------------------------------------------
// Kernel 5: fused MLP:  y += gelu(LN(y) @ w1 + b1) @ w2 + b2, per 64-token
// tile; hidden state (64x384 f16) held in LDS -> no HBM round trip.
// ---------------------------------------------------------------------------
__global__ __launch_bounds__(128) void k_mlp(
    const float* __restrict__ g2, const float* __restrict__ bt2,
    const float* __restrict__ w1, const float* __restrict__ b1,
    const float* __restrict__ w2, const float* __restrict__ b2,
    float* __restrict__ y) {
  __shared__ _Float16 Xs[64][96];
  __shared__ _Float16 T1[64][384];
  const size_t tok0 = (size_t)blockIdx.x * 64;
  {  // LN: two adjacent lanes share one token (48 features each)
    const int pair = threadIdx.x >> 1, hl = threadIdx.x & 1;
    const float* yr = y + (tok0 + pair) * 96 + hl * 48;
    float vbuf[48];
    float s = 0.f, ss = 0.f;
#pragma unroll
    for (int i = 0; i < 48; ++i) { float v = yr[i]; vbuf[i] = v; s += v; ss += v * v; }
    s += __shfl_xor(s, 1, 32);
    ss += __shfl_xor(ss, 1, 32);
    float mu = s * (1.f / 96.f);
    float var = ss * (1.f / 96.f) - mu * mu;
    float rs = rsqrtf(var + 1e-5f);
#pragma unroll
    for (int i = 0; i < 48; ++i) {
      int c = hl * 48 + i;
      Xs[pair][c] = (_Float16)((vbuf[i] - mu) * rs * g2[c] + bt2[c]);
    }
  }
  __syncthreads();
  const int wv = threadIdx.x >> 5, lane = threadIdx.x & 31;
  const int half = (lane >> 4) & 1, cl = lane & 15;
  // GEMM1 (64x96 . 96x384) + bias + tanh-gelu -> T1 (f16, LDS)
#pragma unroll 1
  for (int tile = wv; tile < 96; tile += 4) {
    const int mt = tile / 24, nt = tile % 24;
    v8f acc = {};
#pragma unroll
    for (int ks = 0; ks < 3; ++ks) {
      v16h a = frag_a_f16(&Xs[mt * 16][ks * 32], 96);
      v16h b = frag_b_f32(w1 + (size_t)(ks * 32) * 384 + nt * 16, 384);
      acc = wmma32(a, b, acc);
    }
    const float bb = b1[nt * 16 + cl];
#pragma unroll
    for (int r = 0; r < 8; ++r) {
      float v = acc[r] + bb;
      float gl = 0.5f * v * (1.f + tanhf(0.7978845608028654f * (v + 0.044715f * v * v * v)));
      T1[mt * 16 + half * 8 + r][nt * 16 + cl] = (_Float16)gl;
    }
  }
  __syncthreads();
  // GEMM2 (64x384 . 384x96) + bias + residual, in place on y
#pragma unroll 1
  for (int tile = wv; tile < 24; tile += 4) {
    const int mt = tile / 6, nt = tile % 6;
    v8f acc = {};
#pragma unroll
    for (int ks = 0; ks < 12; ++ks) {
      v16h a = frag_a_f16(&T1[mt * 16][ks * 32], 384);
      v16h b = frag_b_f32(w2 + (size_t)(ks * 32) * 96 + nt * 16, 96);
      acc = wmma32(a, b, acc);
    }
    const float bb = b2[nt * 16 + cl];
#pragma unroll
    for (int r = 0; r < 8; ++r) {
      size_t addr = (tok0 + mt * 16 + half * 8 + r) * 96 + nt * 16 + cl;
      y[addr] = y[addr] + acc[r] + bb;
    }
  }
}

// ---------------------------------------------------------------------------
extern "C" void kernel_launch(void* const* d_in, const int* in_sizes, int n_in,
                              void* d_out, int out_size, void* d_ws, size_t ws_size,
                              hipStream_t stream) {
  (void)in_sizes; (void)n_in; (void)out_size; (void)ws_size;
  const float* x      = (const float*)d_in[0];
  const float* gamma1 = (const float*)d_in[1];
  const float* beta1  = (const float*)d_in[2];
  const float* w_qkv  = (const float*)d_in[3];
  const float* w_out  = (const float*)d_in[4];
  const float* b_out  = (const float*)d_in[5];
  const float* rel_b  = (const float*)d_in[6];
  const float* gamma2 = (const float*)d_in[7];
  const float* beta2  = (const float*)d_in[8];
  const float* w1     = (const float*)d_in[9];
  const float* b1     = (const float*)d_in[10];
  const float* w2     = (const float*)d_in[11];
  const float* b2     = (const float*)d_in[12];
  float* y = (float*)d_out;  // [2,8,128,128,96] f32

  char* ws = (char*)d_ws;
  _Float16* qkv = (_Float16*)ws;                               // 262144*288 f16 (151 MB)
  _Float16* xn  = (_Float16*)(ws + (size_t)262144 * 288 * 2);  // 262144*96 f16 (50 MB)
  _Float16* aout = xn;                                         // reused after k_qkv

  k_ln_part<<<dim3(65536), dim3(128), 0, stream>>>(x, gamma1, beta1, xn);
  k_qkv    <<<dim3(2048),  dim3(128), 0, stream>>>(xn, w_qkv, qkv);
  k_attn   <<<dim3(6144),  dim3(128), 0, stream>>>(qkv, rel_b, aout);
  k_proj   <<<dim3(2048),  dim3(128), 0, stream>>>(aout, w_out, b_out, x, y);
  k_mlp    <<<dim3(4096),  dim3(128), 0, stream>>>(gamma2, beta2, w1, b1, w2, b2, y);
}
